// SAPL_82111184765331
// MI455X (gfx1250) — matrix operand order repaired and checked by reading, not compiled, not run on hardware
//
#include <hip/hip_runtime.h>
#include <hip/hip_bf16.h>
#include <math.h>

// ---------------- problem constants ----------------
#define B_   4
#define N_   2048
#define D_   512
#define H_   8
#define DK_  64
#define KK_  1024          // K_KEEP = N/2
#define BH_  (B_*H_)       // 32
#define ALPHA_ 0.85f
#define PR_ITERS_ 100

// ---------------- gfx1250 async-to-LDS availability ----------------
#if defined(__has_builtin)
#  if __has_builtin(__builtin_amdgcn_global_load_async_to_lds_b128)
#    define HAVE_ASYNC_LDS 1
#  else
#    define HAVE_ASYNC_LDS 0
#  endif
#  if __has_builtin(__builtin_amdgcn_s_wait_asynccnt)
#    define HAVE_WAIT_ASYNC_BUILTIN 1
#  else
#    define HAVE_WAIT_ASYNC_BUILTIN 0
#  endif
#else
#  define HAVE_ASYNC_LDS 0
#  define HAVE_WAIT_ASYNC_BUILTIN 0
#endif

// ---------------- WMMA types / helpers ----------------
typedef __bf16 bf16x16 __attribute__((ext_vector_type(16)));
typedef float  f32x8   __attribute__((ext_vector_type(8)));
typedef int    v4i     __attribute__((ext_vector_type(4)));

union Frag16 { bf16x16 v; uint4 u[2]; };

// A-fragment (16x32 bf16): lane holds row m=lane&15, k = (lane>>4)*16 + e
// => 16 contiguous halves = two contiguous 16B loads.
__device__ inline bf16x16 load_frag_gap16(const unsigned short* p) {
    Frag16 f;
    f.u[0] = *reinterpret_cast<const uint4*>(p);
    f.u[1] = *reinterpret_cast<const uint4*>(p + 8);
    return f.v;
}
// B-fragment (32x16 bf16): lane holds col n=lane&15, k = (lane>>4)*16 + e
// => 16 contiguous halves = two contiguous 16B loads.
__device__ inline bf16x16 load_frag_contig(const unsigned short* p) {
    Frag16 f;
    f.u[0] = *reinterpret_cast<const uint4*>(p);
    f.u[1] = *reinterpret_cast<const uint4*>(p + 8);
    return f.v;
}

__device__ inline f32x8 wmma_bf16(bf16x16 a, bf16x16 b, f32x8 c) {
    // (neg_a, A, neg_b, B, c_mod, C, reuse_a, reuse_b)
    return __builtin_amdgcn_wmma_f32_16x16x32_bf16(false, a, false, b,
                                                   (short)0, c, false, false);
}

__device__ inline unsigned short f2bf(float f) {
    unsigned int u = __float_as_uint(f);
    unsigned int r = (u + 0x7FFFu + ((u >> 16) & 1u)) >> 16;
    return (unsigned short)r;
}

// one 16-byte global -> LDS transfer (async on gfx1250 when available)
__device__ inline void stage_b128(const unsigned short* g, unsigned short* l) {
#if HAVE_ASYNC_LDS
    __builtin_amdgcn_global_load_async_to_lds_b128(
        (__attribute__((address_space(1))) v4i*)g,
        (__attribute__((address_space(3))) v4i*)l, 0, 0);
#else
    *reinterpret_cast<uint4*>(l) = *reinterpret_cast<const uint4*>(g);
#endif
}

__device__ inline void wait_async_lds() {
#if HAVE_ASYNC_LDS
#  if HAVE_WAIT_ASYNC_BUILTIN
    __builtin_amdgcn_s_wait_asynccnt(0);
#  else
    asm volatile("s_wait_asynccnt 0" ::: "memory");
#  endif
#endif
}

// ---------------- elementwise: f32 -> bf16 ----------------
__global__ void k_f32_to_bf16(const float* __restrict__ src,
                              unsigned short* __restrict__ dst, int n) {
    int i = blockIdx.x * blockDim.x + threadIdx.x;
    if (i < n) dst[i] = f2bf(src[i]);
}

// ---------------- positional encoding ----------------
__global__ void k_pos_encoding(float* __restrict__ pe) {
    int i = blockIdx.x * blockDim.x + threadIdx.x;   // over N*(D/2)
    if (i >= N_ * (D_/2)) return;
    int pos = i / (D_/2);
    int j   = i % (D_/2);
    float inv = __expf((float)(2*j) * (-logf(10000.0f) / (float)D_));
    float ang = (float)pos * inv;
    pe[(size_t)pos * D_ + 2*j]     = sinf(ang);
    pe[(size_t)pos * D_ + 2*j + 1] = cosf(ang);
}

// ---------------- LDS-staged bf16 WMMA GEMM: C = A @ W^T (+bias,+pe,relu) ------
// A: [M,K] bf16 row-major, W: [Nout,K] bf16 row-major (so B[k,n] = W[n,k]).
// Block = 8 waves -> 128(M) x 64(N) tile. Weight slab (64 x KC) staged in LDS
// per K-chunk via async-to-LDS; each wave computes a 16x64 strip, reusing each
// A fragment across 4 WMMAs.
#define KC_ 128                      // K-chunk staged per round
#define BPAD_ 8                      // row pad (halves) to break bank conflicts
__global__ __launch_bounds__(256)
void k_gemm_bf16(const unsigned short* __restrict__ A,
                 const unsigned short* __restrict__ Wt,
                 const float* __restrict__ bias,
                 const float* __restrict__ pe,        // nullable: add pe[(row%seqN)*D_+col]
                 float* __restrict__ Cf,              // nullable f32 out [M,Nout]
                 unsigned short* __restrict__ Cb,     // nullable bf16 out [M,Nout]
                 unsigned short* __restrict__ Cqkv,   // nullable bf16 out [B,H,seqN,DK]
                 int M, int K, int Nout, int seqN, int relu)
{
    __shared__ __attribute__((aligned(16))) unsigned short Bs[64][KC_ + BPAD_];

    int lane = threadIdx.x & 31;
    int wv   = threadIdx.x >> 5;
    int nblk = Nout >> 6;                     // N tiles of 64
    int bm = blockIdx.x / nblk;
    int bn = blockIdx.x % nblk;
    int hi = lane >> 4, ln = lane & 15;

    const unsigned short* arow = A + (size_t)(bm*128 + wv*16 + ln) * K;

    f32x8 acc[4];
#pragma unroll
    for (int c = 0; c < 4; ++c) acc[c] = (f32x8){0.f,0.f,0.f,0.f,0.f,0.f,0.f,0.f};

    for (int kc = 0; kc < K; kc += KC_) {
        // cooperatively stage 64 x KC_ weight slab into LDS (async b128)
        for (int v = threadIdx.x; v < 64 * (KC_/8); v += 256) {
            int row = v >> 4;                 // KC_/8 == 16 vectors per row
            int vc  = v & 15;
            stage_b128(Wt + (size_t)(bn*64 + row) * K + kc + vc*8,
                       &Bs[row][vc*8]);
        }
        wait_async_lds();
        __syncthreads();

        // prefetch next A chunk while computing on this one
        if (kc + KC_ < K)
            __builtin_prefetch(arow + kc + KC_, 0, 0);

#pragma unroll
        for (int k0 = 0; k0 < KC_; k0 += 32) {
            bf16x16 a = load_frag_gap16(arow + kc + k0 + hi*16);
#pragma unroll
            for (int c = 0; c < 4; ++c) {
                bf16x16 b = load_frag_contig(&Bs[c*16 + ln][k0 + hi*16]);
                acc[c] = wmma_bf16(a, b, acc[c]);
            }
        }
        __syncthreads();                      // before next slab overwrite
    }

#pragma unroll
    for (int c = 0; c < 4; ++c) {
#pragma unroll
        for (int r = 0; r < 8; ++r) {
            int row = bm*128 + wv*16 + hi*8 + r;
            int col = bn*64 + c*16 + ln;
            float v = acc[c][r];
            if (bias) v += bias[col];
            if (pe)   v += pe[(size_t)(row % seqN) * D_ + col];
            if (relu) v = fmaxf(v, 0.f);
            if (Cf) Cf[(size_t)row * Nout + col] = v;
            if (Cb) Cb[(size_t)row * Nout + col] = f2bf(v);
            if (Cqkv) {
                int b_ = row / seqN, sr = row % seqN;
                int h_ = col >> 6, dk = col & 63;
                Cqkv[(((size_t)b_*H_ + h_) * seqN + sr) * DK_ + dk] = f2bf(v);
            }
        }
    }
}

// ---------------- attention pass 1: per-row online softmax stats ----------------
// Q,Kb: [BH, N, DK] bf16. Output rowmax / rowinv(=1/rowsum): [BH, N] f32.
__global__ __launch_bounds__(256)
void k_attn_rowstats(const unsigned short* __restrict__ Q,
                     const unsigned short* __restrict__ Kb,
                     float* __restrict__ rowmax, float* __restrict__ rowinv)
{
    int lane = threadIdx.x & 31, wv = threadIdx.x >> 5;
    int wid = blockIdx.x * (blockDim.x >> 5) + wv;      // BH * (N/16) waves
    int qt = wid & (N_/16 - 1);
    int bh = wid >> 7;                                  // N/16 == 128 -> shift 7
    int hi = lane >> 4, ln = lane & 15;
    const float scale = 0.125f;                          // 1/sqrt(DK)

    const unsigned short* qrow = Q + ((size_t)bh * N_ + qt*16 + ln) * DK_;
    bf16x16 qa0 = load_frag_gap16(qrow + hi*16);
    bf16x16 qa1 = load_frag_gap16(qrow + 32 + hi*16);

    float mrun[8], srun[8];
#pragma unroll
    for (int r = 0; r < 8; ++r) { mrun[r] = -1e30f; srun[r] = 0.f; }

    for (int kt = 0; kt < N_/16; ++kt) {
        const unsigned short* krow = Kb + ((size_t)bh * N_ + kt*16 + ln) * DK_;
        bf16x16 kb0 = load_frag_contig(krow + hi*16);
        bf16x16 kb1 = load_frag_contig(krow + 32 + hi*16);
        f32x8 s = {0.f,0.f,0.f,0.f,0.f,0.f,0.f,0.f};
        s = wmma_bf16(qa0, kb0, s);
        s = wmma_bf16(qa1, kb1, s);
#pragma unroll
        for (int r = 0; r < 8; ++r) {
            float v = s[r] * scale;
            float tm = v;
            tm = fmaxf(tm, __shfl_xor(tm, 1));
            tm = fmaxf(tm, __shfl_xor(tm, 2));
            tm = fmaxf(tm, __shfl_xor(tm, 4));
            tm = fmaxf(tm, __shfl_xor(tm, 8));
            float nm = fmaxf(mrun[r], tm);
            float e  = __expf(v - nm);
            e += __shfl_xor(e, 1);
            e += __shfl_xor(e, 2);
            e += __shfl_xor(e, 4);
            e += __shfl_xor(e, 8);
            srun[r] = srun[r] * __expf(mrun[r] - nm) + e;
            mrun[r] = nm;
        }
    }
    if (ln == 0) {
#pragma unroll
        for (int r = 0; r < 8; ++r) {
            size_t idx = (size_t)bh * N_ + qt*16 + hi*8 + r;
            rowmax[idx] = mrun[r];
            rowinv[idx] = 1.0f / srun[r];
        }
    }
}

// ---------------- mean_attn: recompute scores, average softmax over 32 (b,h) ----------------
__global__ __launch_bounds__(256)
void k_mean_attn(const unsigned short* __restrict__ Q,
                 const unsigned short* __restrict__ Kb,
                 const float* __restrict__ rowmax, const float* __restrict__ rowinv,
                 float* __restrict__ out)                // [N,N]
{
    int lane = threadIdx.x & 31, wv = threadIdx.x >> 5;
    int wid = blockIdx.x * (blockDim.x >> 5) + wv;       // (N/16)^2 waves
    int kt = wid & (N_/16 - 1);
    int qt = wid >> 7;
    int hi = lane >> 4, ln = lane & 15;
    const float scale = 0.125f;

    float acc[8];
#pragma unroll
    for (int r = 0; r < 8; ++r) acc[r] = 0.f;

    for (int bh = 0; bh < BH_; ++bh) {
        const unsigned short* qrow = Q + ((size_t)bh * N_ + qt*16 + ln) * DK_;
        const unsigned short* krow = Kb + ((size_t)bh * N_ + kt*16 + ln) * DK_;
        bf16x16 qa0 = load_frag_gap16(qrow + hi*16);
        bf16x16 qa1 = load_frag_gap16(qrow + 32 + hi*16);
        bf16x16 kb0 = load_frag_contig(krow + hi*16);
        bf16x16 kb1 = load_frag_contig(krow + 32 + hi*16);
        f32x8 s = {0.f,0.f,0.f,0.f,0.f,0.f,0.f,0.f};
        s = wmma_bf16(qa0, kb0, s);
        s = wmma_bf16(qa1, kb1, s);
#pragma unroll
        for (int r = 0; r < 8; ++r) {
            size_t row = (size_t)bh * N_ + qt*16 + hi*8 + r;
            acc[r] += __expf(s[r]*scale - rowmax[row]) * rowinv[row];
        }
    }
#pragma unroll
    for (int r = 0; r < 8; ++r)
        out[(size_t)(qt*16 + hi*8 + r) * N_ + kt*16 + ln] = acc[r] * (1.0f / BH_);
}

// ---------------- pagerank helpers ----------------
__global__ void k_row_inv_sum(const float* __restrict__ Wm, float* __restrict__ inv) {
    __shared__ float red[256];
    int row = blockIdx.x, t = threadIdx.x;
    float s = 0.f;
    for (int j = t; j < N_; j += 256) s += Wm[(size_t)row * N_ + j];
    red[t] = s; __syncthreads();
    for (int st = 128; st > 0; st >>= 1) {
        if (t < st) red[t] += red[t + st];
        __syncthreads();
    }
    if (t == 0) inv[row] = 1.0f / red[0];
}

__global__ void k_pr_init(float* __restrict__ p) {
    int i = blockIdx.x * blockDim.x + threadIdx.x;
    if (i < N_) p[i] = 1.0f / (float)N_;
}

__global__ void k_pr_iter(const float* __restrict__ Wm, const float* __restrict__ winv,
                          const float* __restrict__ pin, float* __restrict__ pout) {
    __shared__ float ps[N_];
    for (int i = threadIdx.x; i < N_; i += blockDim.x)
        ps[i] = pin[i] * winv[i] * ALPHA_;
    __syncthreads();
    int j = blockIdx.x * blockDim.x + threadIdx.x;
    float acc = (1.0f - ALPHA_) / (float)N_;
    for (int i = 0; i < N_; ++i) {
        if ((i & 15) == 0 && i + 16 < N_)
            __builtin_prefetch(&Wm[(size_t)(i + 16) * N_ + j], 0, 0);
        acc += ps[i] * Wm[(size_t)i * N_ + j];
    }
    pout[j] = acc;
}

// rank-based median (assumes distinct values)
__global__ void k_median(const float* __restrict__ pr, float* __restrict__ median) {
    int i = blockIdx.x * blockDim.x + threadIdx.x;
    if (i >= N_) return;
    float v = pr[i];
    int cnt = 0;
    for (int j = 0; j < N_; ++j) cnt += (pr[j] < v) ? 1 : 0;
    if (cnt == (N_ - 1) / 2) *median = v;
}

__global__ void k_select(const float* __restrict__ pr, const float* __restrict__ median,
                         float* __restrict__ mask_out, int* __restrict__ kept) {
    if (blockIdx.x == 0 && threadIdx.x == 0) {
        float med = *median;
        int c = 0;
        for (int i = 0; i < N_; ++i) {
            int m = (pr[i] <= med) ? 1 : 0;
            mask_out[i] = m ? 1.0f : 0.0f;
            if (!m && c < KK_) kept[c++] = i;
        }
        while (c < KK_) kept[c++] = 0;
    }
}

// ---------------- gather V^T for the P@V WMMA: vkT[bh][d][kk] ----------------
__global__ void k_gather_vT(const unsigned short* __restrict__ Vb, const int* __restrict__ kept,
                            unsigned short* __restrict__ vkT) {
    int i = blockIdx.x * blockDim.x + threadIdx.x;   // B*H*DK*KK
    if (i >= BH_ * DK_ * KK_) return;
    int kk = i % KK_;
    int rest = i / KK_;
    int dk = rest % DK_;
    int bh = rest / DK_;
    vkT[i] = Vb[((size_t)bh * N_ + kept[kk]) * DK_ + dk];
}

// ---------------- attention pass 2: gathered flash P@V -> ao [B,KK,D] ----------------
__global__ __launch_bounds__(256)
void k_attn_out(const unsigned short* __restrict__ Q,
                const unsigned short* __restrict__ Kb,
                const unsigned short* __restrict__ vkT,
                const int* __restrict__ kept,
                const float* __restrict__ rowmax, const float* __restrict__ rowinv,
                float* __restrict__ ao)
{
    __shared__ __attribute__((aligned(16))) unsigned short P[8][16][32];
    int lane = threadIdx.x & 31, wv = threadIdx.x >> 5;
    int wid = blockIdx.x * (blockDim.x >> 5) + wv;     // BH * (KK/16) waves
    int qt = wid & (KK_/16 - 1);
    int bh = wid >> 6;                                  // KK/16 == 64 -> shift 6
    int hi = lane >> 4, ln = lane & 15;
    const float scale = 0.125f;

    int qidx = kept[qt*16 + ln];
    const unsigned short* qrow = Q + ((size_t)bh * N_ + qidx) * DK_;
    bf16x16 qa0 = load_frag_gap16(qrow + hi*16);
    bf16x16 qa1 = load_frag_gap16(qrow + 32 + hi*16);

    float rmax[8], rinv[8];
#pragma unroll
    for (int r = 0; r < 8; ++r) {
        int qr = kept[qt*16 + hi*8 + r];
        size_t idx = (size_t)bh * N_ + qr;
        rmax[r] = rowmax[idx];
        rinv[r] = rowinv[idx];
    }

    f32x8 accv[4];
#pragma unroll
    for (int c = 0; c < 4; ++c) accv[c] = (f32x8){0.f,0.f,0.f,0.f,0.f,0.f,0.f,0.f};

    for (int kt = 0; kt < KK_/32; ++kt) {
        // two 16x16 score tiles covering 32 kept keys
#pragma unroll
        for (int half = 0; half < 2; ++half) {
            int kkey = kept[kt*32 + half*16 + ln];
            const unsigned short* krow = Kb + ((size_t)bh * N_ + kkey) * DK_;
            bf16x16 kb0 = load_frag_contig(krow + hi*16);
            bf16x16 kb1 = load_frag_contig(krow + 32 + hi*16);
            f32x8 s = {0.f,0.f,0.f,0.f,0.f,0.f,0.f,0.f};
            s = wmma_bf16(qa0, kb0, s);
            s = wmma_bf16(qa1, kb1, s);
#pragma unroll
            for (int r = 0; r < 8; ++r) {
                float p = __expf(s[r]*scale - rmax[r]) * rinv[r];
                P[wv][hi*8 + r][half*16 + ln] = f2bf(p);   // C-frag -> LDS tile
            }
        }
        // reload P as a bf16 A-fragment (wave-private LDS slice; no barrier needed)
        const unsigned short* prow = &P[wv][ln][0];
        bf16x16 pa = load_frag_gap16(prow + hi*16);
        // P(16x32) @ V(32x16) over 4 DK chunks
#pragma unroll
        for (int c = 0; c < 4; ++c) {
            const unsigned short* vrow =
                vkT + ((size_t)bh * DK_ + c*16 + ln) * KK_ + kt*32 + hi*16;
            bf16x16 vb = load_frag_contig(vrow);
            accv[c] = wmma_bf16(pa, vb, accv[c]);
        }
    }

    int b_ = bh >> 3, h_ = bh & 7;
#pragma unroll
    for (int c = 0; c < 4; ++c)
#pragma unroll
        for (int r = 0; r < 8; ++r) {
            int qpos = qt*16 + hi*8 + r;
            ao[((size_t)b_*KK_ + qpos) * D_ + h_*DK_ + c*16 + ln] = accv[c][r];
        }
}

// ---------------- layernorm (row = 512, block = 256 threads) ----------------
__global__ __launch_bounds__(256)
void k_layernorm(const float* __restrict__ xa, const float* __restrict__ xb,
                 const int* __restrict__ kept_or_null,   // if set: xa row gathered via kept
                 const float* __restrict__ g, const float* __restrict__ be,
                 float* __restrict__ y, unsigned short* __restrict__ yb)
{
    int row = blockIdx.x;                // B*KK rows
    int t = threadIdx.x;
    const float* arow;
    if (kept_or_null) {
        int b_ = row / KK_, kp = row % KK_;
        arow = xa + ((size_t)b_ * N_ + kept_or_null[kp]) * D_;
    } else {
        arow = xa + (size_t)row * D_;
    }
    const float* brow = xb + (size_t)row * D_;

    __shared__ float red[256];
    float x0 = arow[t]       + brow[t];
    float x1 = arow[t + 256] + brow[t + 256];

    red[t] = x0 + x1; __syncthreads();
    for (int st = 128; st > 0; st >>= 1) { if (t < st) red[t] += red[t + st]; __syncthreads(); }
    float mean = red[0] / (float)D_;
    __syncthreads();

    float d0 = x0 - mean, d1 = x1 - mean;
    red[t] = d0*d0 + d1*d1; __syncthreads();
    for (int st = 128; st > 0; st >>= 1) { if (t < st) red[t] += red[t + st]; __syncthreads(); }
    float rs = rsqrtf(red[0] / (float)D_ + 1e-5f);

    float v0 = d0 * rs * g[t]       + be[t];
    float v1 = d1 * rs * g[t + 256] + be[t + 256];
    y[(size_t)row * D_ + t]       = v0;
    y[(size_t)row * D_ + t + 256] = v1;
    if (yb) {
        yb[(size_t)row * D_ + t]       = f2bf(v0);
        yb[(size_t)row * D_ + t + 256] = f2bf(v1);
    }
}

// ---------------- final pooling: out[b,d] = sum_k pr[kept[k]] * y2[b,k,d] ----------------
__global__ void k_pool(const float* __restrict__ y2, const float* __restrict__ pr,
                       const int* __restrict__ kept, float* __restrict__ out) {
    int i = blockIdx.x * blockDim.x + threadIdx.x;   // B*D
    if (i >= B_ * D_) return;
    int b_ = i / D_, d = i % D_;
    float acc = 0.f;
    for (int k = 0; k < KK_; ++k)
        acc += pr[kept[k]] * y2[((size_t)b_*KK_ + k) * D_ + d];
    out[i] = acc;
}

// ============================================================================
extern "C" void kernel_launch(void* const* d_in, const int* in_sizes, int n_in,
                              void* d_out, int out_size, void* d_ws, size_t ws_size,
                              hipStream_t stream) {
    (void)in_sizes; (void)n_in; (void)out_size; (void)ws_size;
    const float* x    = (const float*)d_in[0];
    const float* w_in = (const float*)d_in[1];
    const float* b_in = (const float*)d_in[2];
    const float* w_q  = (const float*)d_in[3];
    const float* b_q  = (const float*)d_in[4];
    const float* w_k  = (const float*)d_in[5];
    const float* b_k  = (const float*)d_in[6];
    const float* w_v  = (const float*)d_in[7];
    const float* b_v  = (const float*)d_in[8];
    const float* w_f1 = (const float*)d_in[9];
    const float* b_f1 = (const float*)d_in[10];
    const float* w_f2 = (const float*)d_in[11];
    const float* b_f2 = (const float*)d_in[12];
    const float* g1   = (const float*)d_in[13];
    const float* be1  = (const float*)d_in[14];
    const float* g2   = (const float*)d_in[15];
    const float* be2  = (const float*)d_in[16];

    // d_out layout: out [B*D] | mean_attn [N*N] | mask [N]
    float* out_final = (float*)d_out;
    float* mean_attn = out_final + B_*D_;
    float* mask_out  = mean_attn + (size_t)N_*N_;

    // ---- workspace carve-up ----
    char* wp = (char*)d_ws;
    auto alloc = [&](size_t bytes) {
        void* p = wp; wp += (bytes + 255) & ~(size_t)255; return p;
    };
    unsigned short* xb   = (unsigned short*)alloc((size_t)B_*N_*D_*2);
    unsigned short* wib  = (unsigned short*)alloc((size_t)D_*D_*2);
    unsigned short* wqb  = (unsigned short*)alloc((size_t)D_*D_*2);
    unsigned short* wkb  = (unsigned short*)alloc((size_t)D_*D_*2);
    unsigned short* wvb  = (unsigned short*)alloc((size_t)D_*D_*2);
    unsigned short* wf1b = (unsigned short*)alloc((size_t)D_*D_*2);
    unsigned short* wf2b = (unsigned short*)alloc((size_t)D_*D_*2);
    float* pe      = (float*)alloc((size_t)N_*D_*4);
    float* h_f     = (float*)alloc((size_t)B_*N_*D_*4);
    unsigned short* h_b = (unsigned short*)alloc((size_t)B_*N_*D_*2);
    unsigned short* q_b = (unsigned short*)alloc((size_t)B_*N_*D_*2);
    unsigned short* k_b = (unsigned short*)alloc((size_t)B_*N_*D_*2);
    unsigned short* v_b = (unsigned short*)alloc((size_t)B_*N_*D_*2);
    float* rowmax  = (float*)alloc((size_t)BH_*N_*4);
    float* rowinv  = (float*)alloc((size_t)BH_*N_*4);
    float* winv    = (float*)alloc((size_t)N_*4);
    float* p0      = (float*)alloc((size_t)N_*4);
    float* p1      = (float*)alloc((size_t)N_*4);
    float* median  = (float*)alloc(256);
    int*   kept    = (int*)alloc((size_t)KK_*4);
    unsigned short* vkT = (unsigned short*)alloc((size_t)BH_*DK_*KK_*2);
    float* ao      = (float*)alloc((size_t)B_*KK_*D_*4);
    float* y1f     = (float*)alloc((size_t)B_*KK_*D_*4);
    unsigned short* y1b = (unsigned short*)alloc((size_t)B_*KK_*D_*2);
    unsigned short* ff1b= (unsigned short*)alloc((size_t)B_*KK_*D_*2);
    float* ff2f    = (float*)alloc((size_t)B_*KK_*D_*4);
    float* y2f     = (float*)alloc((size_t)B_*KK_*D_*4);

    const int T = 256;

    // 1) f32 -> bf16 conversions
    k_f32_to_bf16<<<(B_*N_*D_ + T-1)/T, T, 0, stream>>>(x, xb, B_*N_*D_);
    k_f32_to_bf16<<<(D_*D_ + T-1)/T, T, 0, stream>>>(w_in, wib, D_*D_);
    k_f32_to_bf16<<<(D_*D_ + T-1)/T, T, 0, stream>>>(w_q,  wqb, D_*D_);
    k_f32_to_bf16<<<(D_*D_ + T-1)/T, T, 0, stream>>>(w_k,  wkb, D_*D_);
    k_f32_to_bf16<<<(D_*D_ + T-1)/T, T, 0, stream>>>(w_v,  wvb, D_*D_);
    k_f32_to_bf16<<<(D_*D_ + T-1)/T, T, 0, stream>>>(w_f1, wf1b, D_*D_);
    k_f32_to_bf16<<<(D_*D_ + T-1)/T, T, 0, stream>>>(w_f2, wf2b, D_*D_);

    // 2) positional encoding
    k_pos_encoding<<<(N_*(D_/2) + T-1)/T, T, 0, stream>>>(pe);

    // 3) h = x @ w_in^T + b_in + pe   (M=8192; block tile 128x64)
    {
        int M = B_*N_, grid = (M/128)*(D_/64);
        k_gemm_bf16<<<grid, T, 0, stream>>>(xb, wib, b_in, pe, h_f, h_b, nullptr,
                                            M, D_, D_, N_, 0);
    }
    // 4) q,k,v GEMMs -> [B,H,N,DK] bf16
    {
        int M = B_*N_, grid = (M/128)*(D_/64);
        k_gemm_bf16<<<grid, T, 0, stream>>>(h_b, wqb, b_q, nullptr, nullptr, nullptr, q_b,
                                            M, D_, D_, N_, 0);
        k_gemm_bf16<<<grid, T, 0, stream>>>(h_b, wkb, b_k, nullptr, nullptr, nullptr, k_b,
                                            M, D_, D_, N_, 0);
        k_gemm_bf16<<<grid, T, 0, stream>>>(h_b, wvb, b_v, nullptr, nullptr, nullptr, v_b,
                                            M, D_, D_, N_, 0);
    }
    // 5) softmax row stats (flash pass 1)
    k_attn_rowstats<<<(BH_*(N_/16))/8, T, 0, stream>>>(q_b, k_b, rowmax, rowinv);
    // 6) mean_attn (recompute pass; q/k are L2-resident)
    k_mean_attn<<<((N_/16)*(N_/16))/8, T, 0, stream>>>(q_b, k_b, rowmax, rowinv, mean_attn);

    // 7) pagerank
    k_row_inv_sum<<<N_, T, 0, stream>>>(mean_attn, winv);
    k_pr_init<<<(N_ + T-1)/T, T, 0, stream>>>(p0);
    for (int it = 0; it < PR_ITERS_; ++it) {
        const float* pin = (it & 1) ? p1 : p0;
        float*       pout= (it & 1) ? p0 : p1;
        k_pr_iter<<<N_/T, T, 0, stream>>>(mean_attn, winv, pin, pout);
    }
    // PR_ITERS even -> final result in p0

    // 8) median, mask, kept
    k_median<<<N_/T, T, 0, stream>>>(p0, median);
    k_select<<<1, 64, 0, stream>>>(p0, median, mask_out, kept);

    // 9) gather V^T (kept keys), then gathered flash P@V
    k_gather_vT<<<(BH_*DK_*KK_)/T, T, 0, stream>>>(v_b, kept, vkT);
    k_attn_out<<<(BH_*(KK_/16))/8, T, 0, stream>>>(q_b, k_b, vkT, kept,
                                                   rowmax, rowinv, ao);

    // 10) y1 = LN(h_k + ao)
    k_layernorm<<<B_*KK_, T, 0, stream>>>(h_f, ao, kept, g1, be1, y1f, y1b);

    // 11) FFN
    {
        int M = B_*KK_, grid = (M/128)*(D_/64);
        k_gemm_bf16<<<grid, T, 0, stream>>>(y1b, wf1b, b_f1, nullptr, nullptr, ff1b, nullptr,
                                            M, D_, D_, N_, 1);
        k_gemm_bf16<<<grid, T, 0, stream>>>(ff1b, wf2b, b_f2, nullptr, ff2f, nullptr, nullptr,
                                            M, D_, D_, N_, 0);
    }
    // 12) y2 = LN(y1 + ff)
    k_layernorm<<<B_*KK_, T, 0, stream>>>(y1f, ff2f, nullptr, g2, be2, y2f, nullptr);

    // 13) out[b,d] = sum_k pr[kept[k]] * y2[b,k,d]
    k_pool<<<(B_*D_)/T, T, 0, stream>>>(y2f, p0, kept, out_final);
}
